// BlockLowRank_76733885710856
// MI455X (gfx1250) — matrix operand order repaired and checked by reading, not compiled, not run on hardware
//
#include <hip/hip_runtime.h>
#include <cstdint>
#include <cstddef>

// ---------------------------------------------------------------------------
// BlockLowRank: out = bias + X @ W^T,  W = einsum('iojr,irjn->iojn').reshape
// Strategy: materialize W once as bf16 hi/lo (64 MB, L2-resident), then
// bf16x3-split WMMA GEMM (v_wmma_f32_16x16x32_bf16) for ~f32 accuracy.
// Block tile 256x128, 8 wave32 in 4x2 grid, wave tile 64x64 (4x4 WMMA tiles).
// W tiles staged with CDNA5 async global->LDS (ASYNCcnt); X converted f32->
// bf16 hi/lo in-registers and stored to LDS.
// ---------------------------------------------------------------------------

typedef __attribute__((ext_vector_type(16))) __bf16 v16bf;
typedef __attribute__((ext_vector_type(8)))  float  v8f;
typedef __attribute__((address_space(3))) unsigned short lds_ushort;

#define IN_DIM  4096
#define OUT_DIM 4096
#define M_TILE  256
#define N_TILE  128
#define K_TILE  32
#define LDSW    40   // padded LDS row stride in ushorts (80B, 16B-aligned)

__device__ __forceinline__ unsigned short f32_to_bf16(float f) {
  unsigned u = __float_as_uint(f);
  u += 0x7FFFu + ((u >> 16) & 1u);          // round-to-nearest-even
  return (unsigned short)(u >> 16);
}
__device__ __forceinline__ float bf16_to_f32(unsigned short h) {
  return __uint_as_float(((unsigned)h) << 16);
}

union FragU { uint4 q[2]; v16bf v; };

// ---------------------------------------------------------------------------
// Kernel 1: W[p][k] = sum_r core1[i,o,j,r] * core2[i,r,j,n]
//   p = i*64+o (out index), k = j*64+n (in index); bf16 hi/lo split output.
// ---------------------------------------------------------------------------
__global__ __launch_bounds__(256) void build_w(
    const float* __restrict__ core1,   // [64,64,64,16]  (i,o,j,r)
    const float* __restrict__ core2,   // [64,16,64,64]  (i,r,j,n)
    unsigned short* __restrict__ w_hi,
    unsigned short* __restrict__ w_lo)
{
  const int idx = blockIdx.x * 256 + threadIdx.x;     // = p*4096 + k
  const int k = idx & (IN_DIM - 1);
  const int p = idx >> 12;
  const int n = k & 63, j = k >> 6;
  const int o = p & 63, i = p >> 6;

  const float* c1 = core1 + ((((size_t)i * 64 + o) * 64 + j) * 16);       // +r
  const float* c2 = core2 + ((size_t)i * 16 * 4096 + (size_t)j * 64 + n); // +r*4096

  float acc = 0.f;
#pragma unroll
  for (int r = 0; r < 16; ++r)
    acc += c1[r] * c2[(size_t)r * 4096];

  const unsigned short hb = f32_to_bf16(acc);
  const unsigned short lb = f32_to_bf16(acc - bf16_to_f32(hb));
  w_hi[idx] = hb;
  w_lo[idx] = lb;
}

// ---------------------------------------------------------------------------
// Kernel 2: Out[m][p] = bias[p] + sum_k X[m][k] * W[p][k]
// ---------------------------------------------------------------------------
__global__ __launch_bounds__(256) void blr_gemm(
    const float* __restrict__ X,              // [M, 4096] f32
    const unsigned short* __restrict__ Whi,   // [4096, 4096] bf16 bits
    const unsigned short* __restrict__ Wlo,   // [4096, 4096] bf16 bits
    const float* __restrict__ bias,           // [4096]
    float* __restrict__ Out)                  // [M, 4096] f32
{
  __shared__ unsigned short lds_xh[M_TILE * LDSW];   // 20 KB
  __shared__ unsigned short lds_xl[M_TILE * LDSW];   // 20 KB
  __shared__ unsigned short lds_wh[N_TILE * LDSW];   // 10 KB
  __shared__ unsigned short lds_wl[N_TILE * LDSW];   // 10 KB

  const int tid  = threadIdx.x;
  const int lane = tid & 31;
  const int wv   = tid >> 5;        // wave id 0..7
  const int wm   = wv >> 1;         // 0..3  (M direction, 64 rows each)
  const int wn   = wv & 1;          // 0..1  (N direction, 64 cols each)
  const int lh   = lane >> 4;       // lane half 0/1
  const int ll   = lane & 15;

  const int bm = blockIdx.y * M_TILE;
  const int bn = blockIdx.x * N_TILE;

  v8f acc[4][4];
  const v8f vzero = {};
#pragma unroll
  for (int a = 0; a < 4; ++a)
#pragma unroll
    for (int b = 0; b < 4; ++b)
      acc[a][b] = vzero;

  // staging thread mapping
  const int xq  = tid & 7;          // float4 slot within 32-col row (8 per row)
  const int xr0 = tid >> 3;         // 0..31  (8 passes of 32 rows -> 256 rows)
  const int wq  = tid & 3;          // uint4 slot within 32-ushort row (4 per row)
  const int wr0 = tid >> 2;         // 0..63  (2 passes of 64 rows -> 128 rows)

  // 32-bit LDS base offsets of the W staging buffers (for async-to-LDS)
  const unsigned ldsb_wh = (unsigned)(uintptr_t)(lds_ushort*)lds_wh;
  const unsigned ldsb_wl = (unsigned)(uintptr_t)(lds_ushort*)lds_wl;

  for (int k0 = 0; k0 < IN_DIM; k0 += K_TILE) {
    // ---- kick off W hi/lo tile staging via async global->LDS (ASYNCcnt) ----
#pragma unroll
    for (int p = 0; p < 2; ++p) {
      const int r = wr0 + p * 64;
      const unsigned goff = (unsigned)(((bn + r) * IN_DIM + k0 + wq * 8) * 2); // bytes
      const unsigned loff = (unsigned)((r * LDSW + wq * 8) * 2);               // bytes
      const unsigned lah = ldsb_wh + loff;
      const unsigned lal = ldsb_wl + loff;
      asm volatile("global_load_async_to_lds_b128 %0, %1, %2 offset:0"
                   :
                   : "v"(lah), "v"(goff), "s"(Whi)
                   : "memory");
      asm volatile("global_load_async_to_lds_b128 %0, %1, %2 offset:0"
                   :
                   : "v"(lal), "v"(goff), "s"(Wlo)
                   : "memory");
    }

    // prefetch next X K-slab toward L2 while we stage/compute this one
    if (k0 + K_TILE < IN_DIM)
      __builtin_prefetch(X + (size_t)(bm + xr0) * IN_DIM + (k0 + K_TILE) + xq * 4, 0, 1);

    // ---- stage X tile: f32 -> bf16 hi/lo into LDS (VALU path) ----
#pragma unroll
    for (int p = 0; p < 8; ++p) {
      const int r = xr0 + p * 32;
      const float4 xv = *(const float4*)(X + (size_t)(bm + r) * IN_DIM + k0 + xq * 4);
      unsigned short h0 = f32_to_bf16(xv.x), h1 = f32_to_bf16(xv.y);
      unsigned short h2 = f32_to_bf16(xv.z), h3 = f32_to_bf16(xv.w);
      unsigned short l0 = f32_to_bf16(xv.x - bf16_to_f32(h0));
      unsigned short l1 = f32_to_bf16(xv.y - bf16_to_f32(h1));
      unsigned short l2 = f32_to_bf16(xv.z - bf16_to_f32(h2));
      unsigned short l3 = f32_to_bf16(xv.w - bf16_to_f32(h3));
      uint2 hv, lv;
      hv.x = (unsigned)h0 | ((unsigned)h1 << 16);
      hv.y = (unsigned)h2 | ((unsigned)h3 << 16);
      lv.x = (unsigned)l0 | ((unsigned)l1 << 16);
      lv.y = (unsigned)l2 | ((unsigned)l3 << 16);
      *(uint2*)&lds_xh[r * LDSW + xq * 4] = hv;
      *(uint2*)&lds_xl[r * LDSW + xq * 4] = lv;
    }

    // async W staging must land before anyone reads LDS after the barrier
    asm volatile("s_wait_asynccnt 0x0" ::: "memory");
    __syncthreads();

    // ---- B fragments: held in registers for the whole K-step ----
    // B 32x16: column n in lane n; lane half 0 holds K 0-15, half 1 K 16-31.
    FragU bh[4], bl[4];
#pragma unroll
    for (int ns = 0; ns < 4; ++ns) {
      const int row = wn * 64 + ns * 16 + ll;
      const int kb  = lh * 16;
      bh[ns].q[0] = *(const uint4*)&lds_wh[row * LDSW + kb];
      bh[ns].q[1] = *(const uint4*)&lds_wh[row * LDSW + kb + 8];
      bl[ns].q[0] = *(const uint4*)&lds_wl[row * LDSW + kb];
      bl[ns].q[1] = *(const uint4*)&lds_wl[row * LDSW + kb + 8];
    }

    // ---- A fragments streamed per ms; 12 WMMAs each ----
    // A 16x32: lane half 0 holds K 0-15; half 1 holds K 16-31.
#pragma unroll
    for (int ms = 0; ms < 4; ++ms) {
      const int row = wm * 64 + ms * 16 + ll;
      const int kb  = lh * 16;
      FragU ah, al;
      ah.q[0] = *(const uint4*)&lds_xh[row * LDSW + kb];
      ah.q[1] = *(const uint4*)&lds_xh[row * LDSW + kb + 8];
      al.q[0] = *(const uint4*)&lds_xl[row * LDSW + kb];
      al.q[1] = *(const uint4*)&lds_xl[row * LDSW + kb + 8];
#pragma unroll
      for (int ns = 0; ns < 4; ++ns) {
        acc[ms][ns] = __builtin_amdgcn_wmma_f32_16x16x32_bf16(
            false, ah.v, false, bh[ns].v, (short)0, acc[ms][ns], false, false);
        acc[ms][ns] = __builtin_amdgcn_wmma_f32_16x16x32_bf16(
            false, ah.v, false, bl[ns].v, (short)0, acc[ms][ns], false, false);
        acc[ms][ns] = __builtin_amdgcn_wmma_f32_16x16x32_bf16(
            false, al.v, false, bh[ns].v, (short)0, acc[ms][ns], false, false);
      }
    }
    __syncthreads();
  }

  // ---- epilogue: C/D layout (VGPR v -> row v + 8*lanehalf), add bias ----
#pragma unroll
  for (int ms = 0; ms < 4; ++ms)
#pragma unroll
    for (int ns = 0; ns < 4; ++ns) {
      const int gr0 = bm + wm * 64 + ms * 16 + lh * 8;
      const int gc  = bn + wn * 64 + ns * 16 + ll;
      const float bv = bias[gc];
#pragma unroll
      for (int v = 0; v < 8; ++v)
        Out[(size_t)(gr0 + v) * OUT_DIM + gc] = acc[ms][ns][v] + bv;
    }
}

// ---------------------------------------------------------------------------
extern "C" void kernel_launch(void* const* d_in, const int* in_sizes, int n_in,
                              void* d_out, int out_size, void* d_ws, size_t ws_size,
                              hipStream_t stream) {
  const float* x     = (const float*)d_in[0];   // [8,2048,4096]
  const float* core1 = (const float*)d_in[1];   // [64,64,64,16]
  const float* core2 = (const float*)d_in[2];   // [64,16,64,64]
  const float* bias  = (const float*)d_in[3];   // [4096]
  float* out = (float*)d_out;

  unsigned short* whi = (unsigned short*)d_ws;                    // 32 MB
  unsigned short* wlo = whi + (size_t)OUT_DIM * IN_DIM;           // 32 MB

  const int M = in_sizes[0] / IN_DIM;           // 16384

  build_w<<<(OUT_DIM * IN_DIM) / 256, 256, 0, stream>>>(core1, core2, whi, wlo);

  dim3 grid(OUT_DIM / N_TILE, M / M_TILE);      // 32 x 64
  blr_gemm<<<grid, 256, 0, stream>>>(x, whi, wlo, bias, out);
}
